// A_NLSOA_49323404427604
// MI455X (gfx1250) — compile-verified, matrix operand order checked by reading, not run on hardware
//
#include <hip/hip_runtime.h>
#include <hip/hip_bf16.h>
#include <cstdint>

// ---------------------------------------------------------------------------
// MI455X (gfx1250). All contractions via v_wmma_f32_16x16x32_bf16.
// 128x128 block tile, 256 threads = 8 waves (2x4 wave grid), 4x2 WMMA tiles
// per wave, K in chunks of 32. Double-buffered LDS: next tile is issued
// (async global->LDS for bf16 copies, global->reg->LDS for fp32->bf16
// conversions) before the WMMA step on the current buffer.
// ---------------------------------------------------------------------------

#define KS 40   // LDS k-stride in bf16 elems: 80B rows (multiple of 16B)
#define TM 128
#define TN 128
#define TK 32

typedef __bf16 bf16_t;
typedef __attribute__((ext_vector_type(16))) __bf16 bf16x16;
typedef __attribute__((ext_vector_type(8)))  float  f32x8;

// ---- CDNA5 async global->LDS (ASYNCcnt-tracked) ---------------------------
// Flat LDS addresses: low 32 bits == LDS byte offset (ISA 10.2 aperture map).
static __device__ __forceinline__ void async_b128(const void* gsrc, const void* ldst) {
  unsigned l = (unsigned)(__SIZE_TYPE__)ldst;
  unsigned long long g = (unsigned long long)(__SIZE_TYPE__)gsrc;
  asm volatile("global_load_async_to_lds_b128 %0, %1, off" :: "v"(l), "v"(g) : "memory");
}
static __device__ __forceinline__ void async_wait0() {
  asm volatile("s_wait_asynccnt 0" ::: "memory");
}

// ---- WMMA fragment load from LDS (16-bit A/B layout, ISA 7.12.2) ----------
static __device__ __forceinline__ bf16x16 load_frag(const bf16_t* lds, int row, int half) {
  union { uint4 u[2]; bf16x16 v; } f;
  const bf16_t* p = lds + row * KS + half * 8;
  f.u[0] = *reinterpret_cast<const uint4*>(p);
  f.u[1] = *reinterpret_cast<const uint4*>(p + 16);
  return f.v;
}

// ---- staging: async bf16 row-tile copy (layout-preserving) ----------------
static __device__ __forceinline__ void issue_a_bf16_async(bf16_t* dst, const bf16_t* A, int lda,
                                                          int m0, int kc, int tid) {
#pragma unroll
  for (int i = 0; i < 2; ++i) {
    int e = tid + i * 256;                    // 512 chunks of 8 bf16 (16B)
    int r = e >> 2, kg = (e & 3) << 3;
    async_b128(A + (size_t)(m0 + r) * lda + kc + kg, dst + r * KS + kg);
  }
}

// ---- staging: fp32 A tile -> bf16 (register pipeline) ---------------------
struct RegA { float4 v[4]; };
static __device__ __forceinline__ void load_a_f32(RegA& r, const float* A, int lda,
                                                  int m0, int kc, int tid) {
#pragma unroll
  for (int i = 0; i < 4; ++i) {
    int e = tid + i * 256;
    int rr = e >> 3, kg = (e & 7) << 2;
    r.v[i] = *reinterpret_cast<const float4*>(A + (size_t)(m0 + rr) * lda + kc + kg);
  }
}
static __device__ __forceinline__ void store_a_f32(const RegA& r, bf16_t* dst, int tid) {
#pragma unroll
  for (int i = 0; i < 4; ++i) {
    int e = tid + i * 256;
    int rr = e >> 3, kg = (e & 7) << 2;
    bf16_t* d = dst + rr * KS + kg;
    d[0] = (bf16_t)r.v[i].x; d[1] = (bf16_t)r.v[i].y;
    d[2] = (bf16_t)r.v[i].z; d[3] = (bf16_t)r.v[i].w;
  }
}

// ---- staging: B [K,N] row-major -> LDS [n][k] (transpose, register pipe) --
struct RegBf { float v[16]; };
static __device__ __forceinline__ void load_b_nn_f32(RegBf& r, const float* Bm, int ldb,
                                                     int n0, int kc, int tid) {
#pragma unroll
  for (int i = 0; i < 16; ++i) {
    int e = tid + i * 256;                    // consecutive tid -> consecutive n (coalesced)
    int k = e >> 7, n = e & 127;
    r.v[i] = Bm[(size_t)(kc + k) * ldb + n0 + n];
  }
}
static __device__ __forceinline__ void store_b_nn_f32(const RegBf& r, bf16_t* dst, int tid) {
#pragma unroll
  for (int i = 0; i < 16; ++i) {
    int e = tid + i * 256;
    int k = e >> 7, n = e & 127;
    dst[n * KS + k] = (bf16_t)r.v[i];
  }
}

struct RegBh { bf16_t v[16]; };
static __device__ __forceinline__ void load_b_nn_bf16(RegBh& r, const bf16_t* Bm, int ldb,
                                                      int n0, int kc, int tid) {
#pragma unroll
  for (int i = 0; i < 16; ++i) {
    int e = tid + i * 256;
    int k = e >> 7, n = e & 127;
    r.v[i] = Bm[(size_t)(kc + k) * ldb + n0 + n];
  }
}
static __device__ __forceinline__ void store_b_nn_bf16(const RegBh& r, bf16_t* dst, int tid) {
#pragma unroll
  for (int i = 0; i < 16; ++i) {
    int e = tid + i * 256;
    int k = e >> 7, n = e & 127;
    dst[n * KS + k] = r.v[i];
  }
}

// ---- core WMMA step: one 32-wide K chunk for this wave's 4x2 tiles --------
static __device__ __forceinline__ void mma_step(const bf16_t* lAs, const bf16_t* lBs,
                                                f32x8 acc[4][2], int waveM, int waveN, int lane) {
  const int half = lane >> 4, l16 = lane & 15;
  bf16x16 af[4];
#pragma unroll
  for (int mt = 0; mt < 4; ++mt)
    af[mt] = load_frag(lAs, waveM * 64 + mt * 16 + l16, half);
  bf16x16 bfr[2];
#pragma unroll
  for (int nt = 0; nt < 2; ++nt)
    bfr[nt] = load_frag(lBs, waveN * 32 + nt * 16 + l16, half);
#pragma unroll
  for (int mt = 0; mt < 4; ++mt)
#pragma unroll
    for (int nt = 0; nt < 2; ++nt)
      acc[mt][nt] = __builtin_amdgcn_wmma_f32_16x16x32_bf16(
          false, af[mt], false, bfr[nt], (short)0, acc[mt][nt], false, false);
}

// ---- generic C = A @ W + bias (W fp32 [K,N]); Cf fp32 (opt accum) / Cb bf16
template <typename AT>
__global__ __launch_bounds__(256) void gemm_w_kernel(
    const AT* __restrict__ A, const float* __restrict__ W,
    const float* __restrict__ bias, float* __restrict__ Cf,
    bf16_t* __restrict__ Cb, int Mm, int Nn, int Kk, int accum) {
  __shared__ __align__(16) bf16_t lAs[2][TM * KS];
  __shared__ __align__(16) bf16_t lBs[2][TN * KS];
  const int tid = threadIdx.x, lane = tid & 31, wid = tid >> 5;
  const int waveM = wid >> 2, waveN = wid & 3;
  const int m0 = blockIdx.y * TM, n0 = blockIdx.x * TN;

  f32x8 acc[4][2];
#pragma unroll
  for (int mt = 0; mt < 4; ++mt)
#pragma unroll
    for (int nt = 0; nt < 2; ++nt) acc[mt][nt] = (f32x8)0.0f;

  // prologue: tile 0 -> buffer 0
  if constexpr (sizeof(AT) == 4) {
    RegA ra; load_a_f32(ra, (const float*)A, Kk, m0, 0, tid); store_a_f32(ra, lAs[0], tid);
  } else {
    issue_a_bf16_async(lAs[0], (const bf16_t*)A, Kk, m0, 0, tid);
  }
  { RegBf rb; load_b_nn_f32(rb, W, Nn, n0, 0, tid); store_b_nn_f32(rb, lBs[0], tid); }
  if constexpr (sizeof(AT) == 2) async_wait0();
  __syncthreads();

  int buf = 0;
  for (int kc = 0; kc < Kk; kc += TK) {
    const bool more = (kc + TK) < Kk;
    RegA ra; RegBf rb;
    if (more) {
      if constexpr (sizeof(AT) == 4)
        load_a_f32(ra, (const float*)A, Kk, m0, kc + TK, tid);
      else
        issue_a_bf16_async(lAs[buf ^ 1], (const bf16_t*)A, Kk, m0, kc + TK, tid);
      load_b_nn_f32(rb, W, Nn, n0, kc + TK, tid);
    }
    mma_step(lAs[buf], lBs[buf], acc, waveM, waveN, lane);
    if (more) {
      if constexpr (sizeof(AT) == 4) store_a_f32(ra, lAs[buf ^ 1], tid);
      store_b_nn_f32(rb, lBs[buf ^ 1], tid);
      if constexpr (sizeof(AT) == 2) async_wait0();
    }
    __syncthreads();
    buf ^= 1;
  }

  const int half = lane >> 4, l16 = lane & 15;
#pragma unroll
  for (int mt = 0; mt < 4; ++mt)
#pragma unroll
    for (int nt = 0; nt < 2; ++nt) {
      int col = n0 + waveN * 32 + nt * 16 + l16;
      float bv = bias ? bias[col] : 0.0f;
#pragma unroll
      for (int v = 0; v < 8; ++v) {
        int row = m0 + waveM * 64 + mt * 16 + half * 8 + v;   // C: M = v + 8*half
        size_t idx = (size_t)row * Nn + col;
        float val = acc[mt][nt][v] + bv;
        if (Cf) Cf[idx] = accum ? (Cf[idx] + val) : val;
        if (Cb) Cb[idx] = (bf16_t)val;
      }
    }
}

// ---- batched logits: L[b,n,m] = (t.t^T)/(h*sqrt(h)) - mu_n*mu_m/sqrt(h) ---
__global__ __launch_bounds__(256) void sigma_kernel(
    const bf16_t* __restrict__ tb, const float* __restrict__ mu,
    float* __restrict__ logits) {
  __shared__ __align__(16) bf16_t lAs[2][TM * KS];
  __shared__ __align__(16) bf16_t lBs[2][TN * KS];
  const int b = blockIdx.z;
  const bf16_t* Tb = tb + (size_t)b * 512 * 1024;
  const float* muB = mu + b * 512;
  float* Lg = logits + (size_t)b * 512 * 512;
  const int tid = threadIdx.x, lane = tid & 31, wid = tid >> 5;
  const int waveM = wid >> 2, waveN = wid & 3;
  const int m0 = blockIdx.y * TM, n0 = blockIdx.x * TN;

  f32x8 acc[4][2];
#pragma unroll
  for (int mt = 0; mt < 4; ++mt)
#pragma unroll
    for (int nt = 0; nt < 2; ++nt) acc[mt][nt] = (f32x8)0.0f;

  issue_a_bf16_async(lAs[0], Tb, 1024, m0, 0, tid);
  issue_a_bf16_async(lBs[0], Tb, 1024, n0, 0, tid);   // NT: B cols == rows of T
  async_wait0();
  __syncthreads();

  int buf = 0;
  for (int kc = 0; kc < 1024; kc += TK) {
    const bool more = (kc + TK) < 1024;
    if (more) {
      issue_a_bf16_async(lAs[buf ^ 1], Tb, 1024, m0, kc + TK, tid);
      issue_a_bf16_async(lBs[buf ^ 1], Tb, 1024, n0, kc + TK, tid);
    }
    mma_step(lAs[buf], lBs[buf], acc, waveM, waveN, lane);
    if (more) async_wait0();
    __syncthreads();
    buf ^= 1;
  }

  const int half = lane >> 4, l16 = lane & 15;
  const float s1 = 1.0f / 32768.0f;   // 1/(h*sqrt(h))
  const float s2 = 1.0f / 32.0f;      // 1/sqrt(h)
#pragma unroll
  for (int mt = 0; mt < 4; ++mt)
#pragma unroll
    for (int nt = 0; nt < 2; ++nt) {
      int col = n0 + waveN * 32 + nt * 16 + l16;
      float mc = muB[col] * s2;
#pragma unroll
      for (int v = 0; v < 8; ++v) {
        int row = m0 + waveM * 64 + mt * 16 + half * 8 + v;
        Lg[(size_t)row * 512 + col] = acc[mt][nt][v] * s1 - muB[row] * mc;
      }
    }
}

// ---- batched ag = att @ g (bf16 NN, bf16 out) -----------------------------
__global__ __launch_bounds__(256) void attg_kernel(
    const bf16_t* __restrict__ att, const bf16_t* __restrict__ g,
    bf16_t* __restrict__ ag) {
  __shared__ __align__(16) bf16_t lAs[2][TM * KS];
  __shared__ __align__(16) bf16_t lBs[2][TN * KS];
  const int b = blockIdx.z;
  const bf16_t* Ab = att + (size_t)b * 512 * 512;
  const bf16_t* Gb = g + (size_t)b * 512 * 1024;
  bf16_t* Ob = ag + (size_t)b * 512 * 1024;
  const int tid = threadIdx.x, lane = tid & 31, wid = tid >> 5;
  const int waveM = wid >> 2, waveN = wid & 3;
  const int m0 = blockIdx.y * TM, n0 = blockIdx.x * TN;

  f32x8 acc[4][2];
#pragma unroll
  for (int mt = 0; mt < 4; ++mt)
#pragma unroll
    for (int nt = 0; nt < 2; ++nt) acc[mt][nt] = (f32x8)0.0f;

  issue_a_bf16_async(lAs[0], Ab, 512, m0, 0, tid);
  { RegBh rb; load_b_nn_bf16(rb, Gb, 1024, n0, 0, tid); store_b_nn_bf16(rb, lBs[0], tid); }
  async_wait0();
  __syncthreads();

  int buf = 0;
  for (int kc = 0; kc < 512; kc += TK) {
    const bool more = (kc + TK) < 512;
    RegBh rb;
    if (more) {
      issue_a_bf16_async(lAs[buf ^ 1], Ab, 512, m0, kc + TK, tid);
      load_b_nn_bf16(rb, Gb, 1024, n0, kc + TK, tid);
    }
    mma_step(lAs[buf], lBs[buf], acc, waveM, waveN, lane);
    if (more) {
      store_b_nn_bf16(rb, lBs[buf ^ 1], tid);
      async_wait0();
    }
    __syncthreads();
    buf ^= 1;
  }

  const int half = lane >> 4, l16 = lane & 15;
#pragma unroll
  for (int mt = 0; mt < 4; ++mt)
#pragma unroll
    for (int nt = 0; nt < 2; ++nt) {
      int col = n0 + waveN * 32 + nt * 16 + l16;
#pragma unroll
      for (int v = 0; v < 8; ++v) {
        int row = m0 + waveM * 64 + mt * 16 + half * 8 + v;
        Ob[(size_t)row * 1024 + col] = (bf16_t)acc[mt][nt][v];
      }
    }
}

// ---- reductions (wave32) ---------------------------------------------------
static __device__ __forceinline__ float warp_red_sum(float v) {
#pragma unroll
  for (int o = 16; o > 0; o >>= 1) v += __shfl_down(v, o, 32);
  return v;
}
static __device__ __forceinline__ float warp_red_max(float v) {
#pragma unroll
  for (int o = 16; o > 0; o >>= 1) v = fmaxf(v, __shfl_down(v, o, 32));
  return v;
}
static __device__ __forceinline__ float block_red_sum(float v, float* sh, int tid, int nwarp) {
  int lane = tid & 31, warp = tid >> 5;
  v = warp_red_sum(v);
  if (lane == 0) sh[warp] = v;
  __syncthreads();
  if (warp == 0) {
    float x = (lane < nwarp) ? sh[lane] : 0.0f;
    x = warp_red_sum(x);
    if (lane == 0) sh[0] = x;
  }
  __syncthreads();
  float r = sh[0];
  __syncthreads();
  return r;
}
static __device__ __forceinline__ float block_red_max(float v, float* sh, int tid, int nwarp) {
  int lane = tid & 31, warp = tid >> 5;
  v = warp_red_max(v);
  if (lane == 0) sh[warp] = v;
  __syncthreads();
  if (warp == 0) {
    float x = (lane < nwarp) ? sh[lane] : -3.402823466e38f;
    x = warp_red_max(x);
    if (lane == 0) sh[0] = x;
  }
  __syncthreads();
  float r = sh[0];
  __syncthreads();
  return r;
}

// ---- BatchNorm1d stats over (B,H) per channel n ---------------------------
__global__ __launch_bounds__(256) void bn_stats_kernel(const float* __restrict__ tpre,
                                                       float* __restrict__ bn_mean,
                                                       float* __restrict__ bn_rstd) {
  __shared__ float sh[8];
  const int n = blockIdx.x, tid = threadIdx.x;
  float s = 0.0f, ss = 0.0f;
  for (int i = tid; i < 64 * 1024; i += 256) {
    int b = i >> 10, h = i & 1023;
    float v = tpre[((size_t)b * 512 + n) * 1024 + h];
    s += v; ss += v * v;
  }
  s = block_red_sum(s, sh, tid, 8);
  ss = block_red_sum(ss, sh, tid, 8);
  if (tid == 0) {
    float mean = s * (1.0f / 65536.0f);
    float var = ss * (1.0f / 65536.0f) - mean * mean;
    bn_mean[n] = mean;
    bn_rstd[n] = rsqrtf(var + 1e-5f);
  }
}

// ---- BN apply + leaky_relu -> bf16 t, plus per-row mean mu[b,n] -----------
__global__ __launch_bounds__(256) void bn_apply_kernel(
    const float* __restrict__ tpre, const float* __restrict__ gamma,
    const float* __restrict__ beta, const float* __restrict__ bn_mean,
    const float* __restrict__ bn_rstd, bf16_t* __restrict__ tb,
    float* __restrict__ mu) {
  __shared__ float sh[8];
  const int row = blockIdx.x, tid = threadIdx.x;
  const int n = row & 511;
  const float ga = gamma[n], be = beta[n], mn = bn_mean[n], rs = bn_rstd[n];
  float sum = 0.0f;
#pragma unroll
  for (int i = 0; i < 4; ++i) {
    int h = tid + i * 256;
    float v = tpre[(size_t)row * 1024 + h];
    v = ga * (v - mn) * rs + be;
    v = (v >= 0.0f) ? v : 0.01f * v;
    tb[(size_t)row * 1024 + h] = (bf16_t)v;
    sum += v;
  }
  sum = block_red_sum(sum, sh, tid, 8);
  if (tid == 0) mu[row] = sum * (1.0f / 1024.0f);
}

// ---- global softmax over flattened 512*512 per batch ----------------------
__global__ __launch_bounds__(1024) void softmax_kernel(const float* __restrict__ logits,
                                                       bf16_t* __restrict__ att) {
  __shared__ float sh[32];
  const int b = blockIdx.x, tid = threadIdx.x;
  const float* L = logits + (size_t)b * 262144;
  bf16_t* O = att + (size_t)b * 262144;
  float mx = -3.402823466e38f;
  for (int i = tid; i < 262144; i += 1024) mx = fmaxf(mx, L[i]);
  mx = block_red_max(mx, sh, tid, 32);
  float s = 0.0f;
  for (int i = tid; i < 262144; i += 1024) s += __expf(L[i] - mx);
  s = block_red_sum(s, sh, tid, 32);
  const float inv = 1.0f / s;
  for (int i = tid; i < 262144; i += 1024) O[i] = (bf16_t)(__expf(L[i] - mx) * inv);
}

// ---------------------------------------------------------------------------
extern "C" void kernel_launch(void* const* d_in, const int* in_sizes, int n_in,
                              void* d_out, int out_size, void* d_ws, size_t ws_size,
                              hipStream_t stream) {
  (void)in_sizes; (void)n_in; (void)out_size; (void)ws_size;
  const float* x     = (const float*)d_in[0];
  const float* W1    = (const float*)d_in[1];
  const float* b1    = (const float*)d_in[2];
  const float* gamma = (const float*)d_in[3];
  const float* beta  = (const float*)d_in[4];
  const float* W2    = (const float*)d_in[5];
  const float* b2    = (const float*)d_in[6];
  const float* W3    = (const float*)d_in[7];
  const float* b3    = (const float*)d_in[8];
  const float* W4    = (const float*)d_in[9];
  const float* b4    = (const float*)d_in[10];
  float* out = (float*)d_out;

  char* w = (char*)d_ws;
  const size_t MB = 1ull << 20;
  float*  t_pre   = (float*)(w);                 // 128 MiB, dead after bn_apply
  float*  logits  = (float*)(w);                 // 64 MiB (aliases dead t_pre)
  bf16_t* att_b   = (bf16_t*)(w + 64 * MB);      // 32 MiB (aliases dead t_pre)
  bf16_t* t_b     = (bf16_t*)(w + 128 * MB);     // 64 MiB
  bf16_t* g_b     = (bf16_t*)(w + 192 * MB);     // 64 MiB
  bf16_t* ag_b    = (bf16_t*)(w + 256 * MB);     // 64 MiB
  float*  mu      = (float*)(w + 320 * MB);      // 128 KiB
  float*  bn_mean = mu + 64 * 512;
  float*  bn_rstd = bn_mean + 512;

  const dim3 blk(256);
  const dim3 gw(1024 / TN, 32768 / TM);          // weight GEMMs: 8 x 256 blocks

  // 1) t_pre = x@W1 + b1 (fp32, needed for BN stats)
  gemm_w_kernel<float><<<gw, blk, 0, stream>>>(x, W1, b1, t_pre, nullptr, 32768, 1024, 1024, 0);
  // 2) g = bf16(x@W2 + b2)
  gemm_w_kernel<float><<<gw, blk, 0, stream>>>(x, W2, b2, nullptr, g_b, 32768, 1024, 1024, 0);
  // 3) phi = x@W4 + b4 -> d_out
  gemm_w_kernel<float><<<gw, blk, 0, stream>>>(x, W4, b4, out, nullptr, 32768, 1024, 1024, 0);
  // 4) BN batch stats per channel n
  bn_stats_kernel<<<dim3(512), blk, 0, stream>>>(t_pre, bn_mean, bn_rstd);
  // 5) BN apply + leaky -> t_b (bf16), row means mu[b,n]
  bn_apply_kernel<<<dim3(32768), blk, 0, stream>>>(t_pre, gamma, beta, bn_mean, bn_rstd, t_b, mu);
  // 6) attention logits: t.t^T/(h*sqrt(h)) - mu⊗mu/sqrt(h)   (t_pre now dead)
  sigma_kernel<<<dim3(512 / TN, 512 / TM, 64), blk, 0, stream>>>(t_b, mu, logits);
  // 7) global softmax per batch -> att_b (bf16)
  softmax_kernel<<<dim3(64), dim3(1024), 0, stream>>>(logits, att_b);
  // 8) ag = att @ g (bf16)
  attg_kernel<<<dim3(1024 / TN, 512 / TM, 64), blk, 0, stream>>>(att_b, g_b, ag_b);
  // 9) out += ag@W3 + b3
  gemm_w_kernel<bf16_t><<<gw, blk, 0, stream>>>(ag_b, W3, b3, out, nullptr, 32768, 1024, 1024, 1);
}
